// Net_936302870909
// MI455X (gfx1250) — matrix operand order repaired and checked by reading, not compiled, not run on hardware
//
#include <hip/hip_runtime.h>
#include <hip/hip_bf16.h>
#include <math.h>

// ---------------------------------------------------------------------------
// 3PU / MPU one-level forward for MI455X (gfx1250, wave32).
//   - KNN Gram matrices + per-channel convs via v_wmma_f32_16x16x32_f16
//   - fused distance + top-k in LDS (never materializes NxN distances in HBM)
//   - feat (an output) is built in-place inside d_out
//   - all zero-padding done branch-free (cndmask) so tile loads stay pipelined
// ---------------------------------------------------------------------------

typedef __attribute__((ext_vector_type(16))) _Float16 v16h;
typedef __attribute__((ext_vector_type(8)))  float    v8f;

#define BATCH 8
#define NPTS  4096
#define NUP   8192
#define KNN   16
#define FINF  3.0e38f

// ------------------------------ normalize ----------------------------------
__global__ __launch_bounds__(256) void normalize_kernel(
    const float* __restrict__ xyz, float* __restrict__ xn,
    float* __restrict__ centroidWs, float* __restrict__ radiusWs,
    float* __restrict__ radiusOut)
{
    __shared__ float red[256];
    __shared__ float cen[3];
    __shared__ float rad;
    const int b = blockIdx.x, t = threadIdx.x;
    const float* xb = xyz + (long)b * 3 * NPTS;

    float s0 = 0.f, s1 = 0.f, s2 = 0.f;
    for (int n = t; n < NPTS; n += 256) {
        s0 += xb[n]; s1 += xb[NPTS + n]; s2 += xb[2 * NPTS + n];
    }
    float sv[3] = { s0, s1, s2 };
    for (int c = 0; c < 3; ++c) {
        red[t] = sv[c]; __syncthreads();
        for (int o = 128; o > 0; o >>= 1) { if (t < o) red[t] += red[t + o]; __syncthreads(); }
        if (t == 0) cen[c] = red[0] / (float)NPTS;
        __syncthreads();
    }
    float mx = 0.f;
    for (int n = t; n < NPTS; n += 256) {
        float dx = xb[n] - cen[0];
        float dy = xb[NPTS + n] - cen[1];
        float dz = xb[2 * NPTS + n] - cen[2];
        mx = fmaxf(mx, dx * dx + dy * dy + dz * dz);
    }
    red[t] = mx; __syncthreads();
    for (int o = 128; o > 0; o >>= 1) { if (t < o) red[t] = fmaxf(red[t], red[t + o]); __syncthreads(); }
    if (t == 0) {
        float r = sqrtf(red[0]);
        rad = r;
        radiusWs[b] = r;
        radiusOut[b] = r;
        centroidWs[b * 3 + 0] = cen[0];
        centroidWs[b * 3 + 1] = cen[1];
        centroidWs[b * 3 + 2] = cen[2];
    }
    __syncthreads();
    const float inv = 1.f / rad;
    for (int n = t; n < NPTS; n += 256) {
        xn[(long)b * 3 * NPTS + 0 * NPTS + n] = (xb[n] - cen[0]) * inv;
        xn[(long)b * 3 * NPTS + 1 * NPTS + n] = (xb[NPTS + n] - cen[1]) * inv;
        xn[(long)b * 3 * NPTS + 2 * NPTS + n] = (xb[2 * NPTS + n] - cen[2]) * inv;
    }
}

// ------------------------------ layer0 (3->24) -----------------------------
__global__ __launch_bounds__(256) void layer0_kernel(
    const float* __restrict__ xn, const float* __restrict__ W,
    const float* __restrict__ bias, float* __restrict__ feat)
{
    const long gid = (long)blockIdx.x * 256 + threadIdx.x;
    const int b = (int)(gid / NPTS), n = (int)(gid % NPTS);
    const float p0 = xn[(long)b * 3 * NPTS + n];
    const float p1 = xn[(long)b * 3 * NPTS + NPTS + n];
    const float p2 = xn[(long)b * 3 * NPTS + 2 * NPTS + n];
    float* fb = feat + (long)b * 264 * NPTS;
#pragma unroll
    for (int o = 0; o < 24; ++o) {
        float v = W[o * 3] * p0 + W[o * 3 + 1] * p1 + W[o * 3 + 2] * p2 + bias[o];
        fb[(long)(240 + o) * NPTS + n] = v;
    }
}

// ------------------------------ squared norms ------------------------------
__global__ __launch_bounds__(256) void sq_kernel(
    const float* __restrict__ x, long xBatchStride, float* __restrict__ sq)
{
    const long gid = (long)blockIdx.x * 256 + threadIdx.x;
    const int b = (int)(gid / NPTS), n = (int)(gid % NPTS);
    const float* xb = x + (long)b * xBatchStride;
    float s = 0.f;
#pragma unroll
    for (int c = 0; c < 24; ++c) {
        float v = xb[(long)c * NPTS + n];
        s += v * v;
    }
    sq[(long)b * NPTS + n] = s;
}

// ------------------------------ KNN (WMMA Gram + fused top-k) --------------
// 16 waves = one 16-row tile. Per chunk each wave computes a 16x32 column
// slab (two v_wmma_f32_16x16x32_f16 tiles) into a 16x512 LDS dist tile; wave
// w then updates per-lane sorted top-16 lists for row w. Zero-padding of the
// 24->32 K dim is branch-free (valid loads always issue; pads via cndmask).
// Final 32-way merge via shfl_xor min-extraction.
__global__ __launch_bounds__(512) void knn_kernel(
    const float* __restrict__ x, long xBatchStride,
    const float* __restrict__ sq, int* __restrict__ idxOut)
{
    __shared__ float ldsDist[16 * 512];      // 32 KB
    __shared__ float ldsSqRow[16];

    const int wave = threadIdx.x >> 5, lane = threadIdx.x & 31;
    const int b = blockIdx.x >> 8;           // NPTS/16 = 256 tiles per batch
    const int i0 = (blockIdx.x & 255) << 4;
    const float* xb = x + (long)b * xBatchStride;
    const float* sqb = sq + (long)b * NPTS;
    const int lh = lane & 15;
    const bool hi = lane >= 16;

    if (threadIdx.x < 16) ldsSqRow[threadIdx.x] = sqb[i0 + threadIdx.x];

    // A tile: 16 query rows x 32 channels (pad 24->32), f16, branch-free.
    v16h a;
    {
        const int ri = i0 + lh;
#pragma unroll
        for (int e = 0; e < 8; ++e) {
            // elements 0..7: channels (hi?8:0)+e, always valid (<=15)
            a[e] = (_Float16)xb[(long)((hi ? 8 : 0) + e) * NPTS + ri];
            // elements 8..15: lanes<16 need ch 16..23; lanes>=16 pad (24..31)
            const float v = xb[(long)(16 + e) * NPTS + ri];
            a[e + 8] = hi ? (_Float16)0.f : (_Float16)v;
        }
    }

    float dlist[KNN]; int jlist[KNN];
#pragma unroll
    for (int t = 0; t < KNN; ++t) { dlist[t] = FINF; jlist[t] = 0; }

    for (int chunk = 0; chunk < 8; ++chunk) {
        __syncthreads();                     // previous selection done
#pragma unroll
        for (int sub = 0; sub < 2; ++sub) {
            const int jcol = chunk * 512 + wave * 32 + sub * 16 + lh;
            v16h bm;
#pragma unroll
            for (int e = 0; e < 8; ++e) {
                bm[e] = (_Float16)xb[(long)((hi ? 8 : 0) + e) * NPTS + jcol];
                const float v = xb[(long)(16 + e) * NPTS + jcol]; // ch 16..23
                bm[e + 8] = hi ? (_Float16)0.f : (_Float16)v;
            }
            v8f acc = {};
            acc = __builtin_amdgcn_wmma_f32_16x16x32_f16(false, a, false, bm,
                                                         (short)0, acc, false, false);
            const float sqc = sqb[jcol];
#pragma unroll
            for (int r = 0; r < 8; ++r) {
                const int m = r + (hi ? 8 : 0);
                float d = ldsSqRow[m] + sqc - 2.f * acc[r];
                if (jcol == i0 + m) d = FINF;     // exclude self (ref drops it)
                ldsDist[m * 512 + (wave * 32 + sub * 16 + lh)] = d;
            }
        }
        __syncthreads();
        // selection: wave handles row `wave`, lanes stride the 512 columns
#pragma unroll
        for (int s = 0; s < 16; ++s) {
            const int cc = lane + s * 32;
            const float d = ldsDist[wave * 512 + cc];
            const int gj = chunk * 512 + cc;
            if (d < dlist[KNN - 1]) {            // unrolled sorted insertion
                dlist[KNN - 1] = d; jlist[KNN - 1] = gj;
#pragma unroll
                for (int t = KNN - 1; t > 0; --t) {
                    if (dlist[t] < dlist[t - 1]) {
                        float td = dlist[t]; dlist[t] = dlist[t - 1]; dlist[t - 1] = td;
                        int   tj = jlist[t]; jlist[t] = jlist[t - 1]; jlist[t - 1] = tj;
                    }
                }
            }
        }
    }

    // merge 32 sorted per-lane lists: 16 rounds of wave-wide min extraction
    const int row = i0 + wave;
    int* orow = idxOut + ((long)b * NPTS + row) * KNN;
    for (int r = 0; r < KNN; ++r) {
        float d = dlist[0]; int jj = jlist[0]; int ln = lane;
#pragma unroll
        for (int off = 16; off > 0; off >>= 1) {
            const float od = __shfl_xor(d, off, 32);
            const int   oj = __shfl_xor(jj, off, 32);
            const int   ol = __shfl_xor(ln, off, 32);
            if (od < d || (od == d && ol < ln)) { d = od; jj = oj; ln = ol; }
        }
        if (lane == 0) orow[r] = jj;
        if (lane == ln) {                        // winner pops head (static shifts)
#pragma unroll
            for (int t = 0; t < KNN - 1; ++t) { dlist[t] = dlist[t + 1]; jlist[t] = jlist[t + 1]; }
            dlist[KNN - 1] = FINF;
        }
    }
}

// ------------------------------ DenseEdgeConv ------------------------------
// Per-thread (one point). Weights broadcast from LDS; k-invariant W*xc terms
// hoisted out of the neighbor loop. Output 60 channels: [h2|relu h1|relu h0|xc].
__global__ __launch_bounds__(256) void edge_conv_kernel(
    const float* __restrict__ x24, long xBatchStride,
    const int* __restrict__ idx,
    const float* __restrict__ W0, const float* __restrict__ b0,
    const float* __restrict__ W1, const float* __restrict__ b1,
    const float* __restrict__ W2, const float* __restrict__ b2,
    float* __restrict__ outFeat, long outBatchStride)
{
    __shared__ float w0s[12 * 48], w1s[12 * 36], w2s[12 * 48];
    __shared__ float b0s[12], b1s[12], b2s[12];
    for (int i = threadIdx.x; i < 576; i += 256) w0s[i] = W0[i];
    for (int i = threadIdx.x; i < 432; i += 256) w1s[i] = W1[i];
    for (int i = threadIdx.x; i < 576; i += 256) w2s[i] = W2[i];
    if (threadIdx.x < 12) {
        b0s[threadIdx.x] = b0[threadIdx.x];
        b1s[threadIdx.x] = b1[threadIdx.x];
        b2s[threadIdx.x] = b2[threadIdx.x];
    }
    __syncthreads();

    const long gid = (long)blockIdx.x * 256 + threadIdx.x;
    const int b = (int)(gid / NPTS), n = (int)(gid % NPTS);
    const float* xb = x24 + (long)b * xBatchStride;

    float xc[24];
#pragma unroll
    for (int c = 0; c < 24; ++c) xc[c] = xb[(long)c * NPTS + n];

    float h0b[12], h1b[12], h2b[12];
#pragma unroll
    for (int o = 0; o < 12; ++o) {
        float s0 = b0s[o], s1 = b1s[o], s2 = b2s[o];
#pragma unroll
        for (int c = 0; c < 24; ++c) {
            s0 += w0s[o * 48 + c]      * xc[c];   // mlp0 input ch c    = xc
            s1 += w1s[o * 36 + 12 + c] * xc[c];   // mlp1 input ch 12+c = xc
            s2 += w2s[o * 48 + 24 + c] * xc[c];   // mlp2 input ch 24+c = xc
        }
        h0b[o] = s0; h1b[o] = s1; h2b[o] = s2;
    }

    float mx[36];
#pragma unroll
    for (int t = 0; t < 36; ++t) mx[t] = -FINF;

    const int* irow = idx + gid * KNN;
    for (int k = 0; k < KNN; ++k) {
        const int j = irow[k];
        float e[24];
#pragma unroll
        for (int c = 0; c < 24; ++c) e[c] = xb[(long)c * NPTS + j] - xc[c];

        float h0[12];
#pragma unroll
        for (int o = 0; o < 12; ++o) {
            float s = h0b[o];
#pragma unroll
            for (int c = 0; c < 24; ++c) s += w0s[o * 48 + 24 + c] * e[c];
            h0[o] = fmaxf(s, 0.f);
        }
        float h1[12];
#pragma unroll
        for (int o = 0; o < 12; ++o) {
            float s = h1b[o];
#pragma unroll
            for (int c = 0; c < 12; ++c) s += w1s[o * 36 + c] * h0[c];
            h1[o] = fmaxf(s, 0.f);
        }
        float h2[12];
#pragma unroll
        for (int o = 0; o < 12; ++o) {
            float s = h2b[o];
#pragma unroll
            for (int c = 0; c < 12; ++c) s += w2s[o * 48 + c] * h1[c];
#pragma unroll
            for (int c = 0; c < 12; ++c) s += w2s[o * 48 + 12 + c] * h0[c];
            h2[o] = s;                            // last mlp: no relu
        }
#pragma unroll
        for (int t = 0; t < 12; ++t) {
            mx[t]      = fmaxf(mx[t], h2[t]);
            mx[12 + t] = fmaxf(mx[12 + t], h1[t]);
            mx[24 + t] = fmaxf(mx[24 + t], h0[t]);
        }
    }

    float* ob = outFeat + (long)b * outBatchStride;
#pragma unroll
    for (int t = 0; t < 36; ++t) ob[(long)t * NPTS + n] = mx[t];
#pragma unroll
    for (int c = 0; c < 24; ++c) ob[(long)(36 + c) * NPTS + n] = xc[c];
}

// ------------------------------ generic WMMA 1x1 conv ----------------------
// out(b,o,n) = act(sum_c W[o,c]*in(b,c,n) + bias[o]).
// mode 0: in layout (B, cin, npos). mode 1 (up1): c<264 reads feat at n>>1,
// c==264 is the 1D code (+-0.2). All OOB padding is clamp-load + cndmask
// (no exec-mask branches around loads).
__global__ __launch_bounds__(128) void wmma_conv_kernel(
    const float* __restrict__ in, const float* __restrict__ W,
    const float* __restrict__ bias, float* __restrict__ out,
    int cin, int cout, int npos, long inBatchStride, long outBatchStride,
    int relu, int mode)
{
    const int wave = threadIdx.x >> 5, lane = threadIdx.x & 31;
    const int ntiles4 = npos >> 6;
    const int cot = (cout + 15) >> 4;
    const int bid = blockIdx.x;
    const int b   = bid / (cot * ntiles4);
    const int rem = bid % (cot * ntiles4);
    const int o0  = (rem / ntiles4) * 16;
    const int n0  = ((rem % ntiles4) * 4 + wave) * 16;
    const float* inB = in + (long)b * inBatchStride;
    const int lh = lane & 15;
    const bool hi = lane >= 16;
    const int cinm1 = cin - 1;

    const int o  = o0 + lh;
    const int oc = (o < cout) ? o : (cout - 1);   // clamped row for loads
    const bool ov = (o < cout);
    const int n  = n0 + lh;

    v8f acc = {};
#pragma unroll 2
    for (int c0 = 0; c0 < cin; c0 += 32) {
        v16h a, bm;
#pragma unroll
        for (int e = 0; e < 8; ++e) {
            const int ka = c0 + (hi ? 16 : 0) + e;
            const int kb = c0 + (hi ? 24 : 8) + e;
            const int kac = (ka <= cinm1) ? ka : cinm1;
            const int kbc = (kb <= cinm1) ? kb : cinm1;
            const float wa = W[(long)oc * cin + kac];
            const float wb = W[(long)oc * cin + kbc];
            a[e]     = (ov && ka <= cinm1) ? (_Float16)wa : (_Float16)0.f;
            a[e + 8] = (ov && kb <= cinm1) ? (_Float16)wb : (_Float16)0.f;
        }
#pragma unroll
        for (int e = 0; e < 16; ++e) {
            const int c = c0 + (hi ? 16 : 0) + e;
            float v;
            if (mode == 0) {
                const int cc = (c <= cinm1) ? c : cinm1;
                const float lv = inB[(long)cc * npos + n];
                v = (c <= cinm1) ? lv : 0.f;
            } else {
                const int cc = (c <= 263) ? c : 263;
                const float lv = inB[(long)cc * NPTS + (n >> 1)];
                const float code = (n & 1) ? 0.2f : -0.2f;
                v = (c < 264) ? lv : ((c == 264) ? code : 0.f);
            }
            bm[e] = (_Float16)v;
        }
        acc = __builtin_amdgcn_wmma_f32_16x16x32_f16(false, a, false, bm,
                                                     (short)0, acc, false, false);
    }

    float* outB = out + (long)b * outBatchStride;
#pragma unroll
    for (int r = 0; r < 8; ++r) {
        const int oo = o0 + r + (hi ? 8 : 0);
        if (oo < cout) {
            float v = acc[r] + bias[oo];
            if (relu) v = fmaxf(v, 0.f);
            outB[(long)oo * npos + (n0 + lh)] = v;
        }
    }
}

// ------------------------------ fc2 + final compose ------------------------
__global__ __launch_bounds__(256) void fc2_final_kernel(
    const float* __restrict__ fin, const float* __restrict__ W,
    const float* __restrict__ bias, const float* __restrict__ xn,
    const float* __restrict__ centroid, const float* __restrict__ radius,
    float* __restrict__ out)
{
    const long gid = (long)blockIdx.x * 256 + threadIdx.x;
    const int b = (int)(gid / NUP), n = (int)(gid % NUP);
    const float* fb = fin + (long)b * 64 * NUP;
    float a0 = bias[0], a1 = bias[1], a2 = bias[2];
#pragma unroll
    for (int k = 0; k < 64; ++k) {
        const float v = fb[(long)k * NUP + n];
        a0 += W[k] * v;
        a1 += W[64 + k] * v;
        a2 += W[128 + k] * v;
    }
    const float r = radius[b];
    const int ns = n >> 1;
    const float x0 = xn[(long)b * 3 * NPTS + ns];
    const float x1 = xn[(long)b * 3 * NPTS + NPTS + ns];
    const float x2 = xn[(long)b * 3 * NPTS + 2 * NPTS + ns];
    out[(long)b * 3 * NUP + n]           = (a0 + x0) * r + centroid[b * 3 + 0];
    out[(long)b * 3 * NUP + NUP + n]     = (a1 + x1) * r + centroid[b * 3 + 1];
    out[(long)b * 3 * NUP + 2 * NUP + n] = (a2 + x2) * r + centroid[b * 3 + 2];
}

// ---------------------------------------------------------------------------
extern "C" void kernel_launch(void* const* d_in, const int* in_sizes, int n_in,
                              void* d_out, int out_size, void* d_ws, size_t ws_size,
                              hipStream_t stream)
{
    (void)in_sizes; (void)n_in; (void)out_size; (void)ws_size;
    // jax tree-leaf order (sorted dict keys):
    // [0] batch_radius (unused by reference math)
    // [1..24] dec1..dec4, each {W0,b0,W1,b1,W2,b2}
    // [25] fc1.W [26] fc1.b [27] fc2.W [28] fc2.b
    // [29] layer0.W [30] layer0.b
    // [31] prep2.W [32] prep2.b [33] prep3.W [34] prep3.b [35] prep4.W [36] prep4.b
    // [37] up1.W [38] up1.b [39] up2.W [40] up2.b
    // [41] ratio (==2, hardcoded) [42] xyz
    const float* xyz = (const float*)d_in[42];
    const float* decW[4][3]; const float* decB[4][3];
    for (int d = 0; d < 4; ++d)
        for (int m = 0; m < 3; ++m) {
            decW[d][m] = (const float*)d_in[1 + d * 6 + m * 2];
            decB[d][m] = (const float*)d_in[1 + d * 6 + m * 2 + 1];
        }
    const float* fc1W = (const float*)d_in[25]; const float* fc1b = (const float*)d_in[26];
    const float* fc2W = (const float*)d_in[27]; const float* fc2b = (const float*)d_in[28];
    const float* l0W  = (const float*)d_in[29]; const float* l0b  = (const float*)d_in[30];
    const float* prW[3] = { (const float*)d_in[31], (const float*)d_in[33], (const float*)d_in[35] };
    const float* prB[3] = { (const float*)d_in[32], (const float*)d_in[34], (const float*)d_in[36] };
    const float* up1W = (const float*)d_in[37]; const float* up1b = (const float*)d_in[38];
    const float* up2W = (const float*)d_in[39]; const float* up2b = (const float*)d_in[40];

    // outputs: [out 8*3*8192][feat 8*264*4096][radius 8]
    float* outMain = (float*)d_out;
    float* feat    = outMain + (long)BATCH * 3 * NUP;
    float* radOut  = feat + (long)BATCH * 264 * NPTS;

    // workspace (float offsets)
    float* ws     = (float*)d_ws;
    float* xn     = ws;                                    // 98304
    float* cur24  = xn + (long)BATCH * 3 * NPTS;           // 786432
    float* sqbuf  = cur24 + (long)BATCH * 24 * NPTS;       // 32768
    float* cenWs  = sqbuf + (long)BATCH * NPTS;            // 24
    float* radWs  = cenWs + 24;                            // 8
    int*   idxWs  = (int*)(radWs + 8);                     // 524288 ints
    float* bufA   = (float*)(idxWs) + (long)BATCH * NPTS * KNN;  // 8388608
    float* bufB   = bufA + (long)BATCH * 128 * NUP;              // 8388608

    const long featB = (long)264 * NPTS;

    normalize_kernel<<<BATCH, 256, 0, stream>>>(xyz, xn, cenWs, radWs, radOut);
    layer0_kernel<<<(BATCH * NPTS) / 256, 256, 0, stream>>>(xn, l0W, l0b, feat);

    // channel bases: dec1->180, dec2->120, dec3->60, dec4->0; x24 at 240.
    const int decBase[4] = { 180, 120, 60, 0 };
    for (int s = 0; s < 4; ++s) {
        const float* x24; long x24B;
        if (s == 0) { x24 = feat + (long)240 * NPTS; x24B = featB; }
        else {
            const int cin = (s == 1) ? 84 : (s == 2) ? 144 : 204;
            const float* tail = feat + (long)decBase[s - 1] * NPTS;
            const int blocks = BATCH * 2 /*cout tiles*/ * (NPTS >> 6);
            wmma_conv_kernel<<<blocks, 128, 0, stream>>>(
                tail, prW[s - 1], prB[s - 1], cur24,
                cin, 24, NPTS, featB, (long)24 * NPTS, 1, 0);
            x24 = cur24; x24B = (long)24 * NPTS;
        }
        sq_kernel<<<(BATCH * NPTS) / 256, 256, 0, stream>>>(x24, x24B, sqbuf);
        knn_kernel<<<BATCH * (NPTS / 16), 512, 0, stream>>>(x24, x24B, sqbuf, idxWs);
        edge_conv_kernel<<<(BATCH * NPTS) / 256, 256, 0, stream>>>(
            x24, x24B, idxWs,
            decW[s][0], decB[s][0], decW[s][1], decB[s][1], decW[s][2], decB[s][2],
            feat + (long)decBase[s] * NPTS, featB);
    }

    // up path (ratio==2): up1 265->128, up2 128->128, fc1 128->64, fc2 64->3
    wmma_conv_kernel<<<BATCH * 8 * (NUP >> 6), 128, 0, stream>>>(
        feat, up1W, up1b, bufA, 265, 128, NUP, featB, (long)128 * NUP, 1, 1);
    wmma_conv_kernel<<<BATCH * 8 * (NUP >> 6), 128, 0, stream>>>(
        bufA, up2W, up2b, bufB, 128, 128, NUP, (long)128 * NUP, (long)128 * NUP, 1, 0);
    wmma_conv_kernel<<<BATCH * 4 * (NUP >> 6), 128, 0, stream>>>(
        bufB, fc1W, fc1b, bufA, 128, 64, NUP, (long)128 * NUP, (long)64 * NUP, 1, 0);
    fc2_final_kernel<<<(BATCH * NUP) / 256, 256, 0, stream>>>(
        bufA, fc2W, fc2b, xn, cenWs, radWs, outMain);
}
